// MultiHeadAttention_12257836663023
// MI455X (gfx1250) — matrix operand order repaired and checked by reading, not compiled, not run on hardware
//
#include <hip/hip_runtime.h>

// ---------------------------------------------------------------------------
// CDNA5 (gfx1250) bf16 WMMA implementation of causal multi-head attention.
// D = A x B + C with V_WMMA_F32_16X16X32_BF16, wave32.
// ---------------------------------------------------------------------------

typedef __bf16 bf16;
typedef bf16 bf16x4 __attribute__((ext_vector_type(4)));
typedef bf16 bf16x8 __attribute__((ext_vector_type(8)));
typedef bf16 v16bf  __attribute__((ext_vector_type(16)));
typedef float v8f   __attribute__((ext_vector_type(8)));

#define DEVI __device__ __forceinline__

// ds_swizzle xor-shuffle within 32 lanes (group-of-32 mode: and=0x1f, xor=m)
#define SWZ_XOR(v, m) \
  __int_as_float(__builtin_amdgcn_ds_swizzle(__float_as_int(v), ((m) << 10) | 0x1f))

DEVI float redmax16(float v) {
  v = fmaxf(v, SWZ_XOR(v, 1));
  v = fmaxf(v, SWZ_XOR(v, 2));
  v = fmaxf(v, SWZ_XOR(v, 4));
  v = fmaxf(v, SWZ_XOR(v, 8));
  return v;
}
DEVI float redsum16(float v) {
  v += SWZ_XOR(v, 1);
  v += SWZ_XOR(v, 2);
  v += SWZ_XOR(v, 4);
  v += SWZ_XOR(v, 8);
  return v;
}

// Wave-local LDS ordering: CDNA5 executes DS ops of one wave in order, so a
// cross-lane LDS write->read within a wave only needs a compiler fence to
// prevent reordering (no block barrier, no hardware wait beyond the load's
// own DScnt which the compiler inserts).
DEVI void wave_lds_fence() {
  __builtin_amdgcn_wave_barrier();
  asm volatile("" ::: "memory");
}

// A fragment (16x32 bf16, MxK). ISA layout: lane<16 -> row M=lane,
// elements = K {0..7, 16..23}; lane>=16 -> row M=lane-16, K {8..15, 24..31}.
// Two 16B loads per lane.
DEVI v16bf ldfragA(const bf16* __restrict__ p, int ld) {
  const int lane = threadIdx.x & 31;
  const int r = lane & 15, hi = lane >> 4;
  const bf16* q = p + r * ld + hi * 8;
  bf16x8 h0 = *(const bf16x8*)(q);
  bf16x8 h1 = *(const bf16x8*)(q + 16);
  return __builtin_shufflevector(h0, h1, 0, 1, 2, 3, 4, 5, 6, 7,
                                 8, 9, 10, 11, 12, 13, 14, 15);
}
DEVI v16bf ldfragA_lds(const bf16* p, int ld) {
  const int lane = threadIdx.x & 31;
  const int r = lane & 15, hi = lane >> 4;
  const bf16* q = p + r * ld + hi * 8;
  bf16x8 h0 = *(const bf16x8*)(q);
  bf16x8 h1 = *(const bf16x8*)(q + 16);
  return __builtin_shufflevector(h0, h1, 0, 1, 2, 3, 4, 5, 6, 7,
                                 8, 9, 10, 11, 12, 13, 14, 15);
}

// B fragment (32x16 bf16, KxN), source tile stored as 16 N-rows of
// contiguous K. Layout: lane holds N=lane&15, K=(lane>>4)*8 + {0..7, 16..23}.
DEVI v16bf ldfragB(const bf16* __restrict__ p, int ld) {
  const int lane = threadIdx.x & 31;
  const int n = lane & 15, ks = lane >> 4;
  const bf16* q = p + n * ld + ks * 8;
  bf16x8 h0 = *(const bf16x8*)(q);
  bf16x8 h1 = *(const bf16x8*)(q + 16);
  return __builtin_shufflevector(h0, h1, 0, 1, 2, 3, 4, 5, 6, 7,
                                 8, 9, 10, 11, 12, 13, 14, 15);
}

DEVI v8f wmma_bf16(v16bf a, v16bf b, v8f c) {
  return __builtin_amdgcn_wmma_f32_16x16x32_bf16(false, a, false, b,
                                                 (short)0, c, false, false);
}

// ---------------------------------------------------------------------------
// f32 -> bf16 conversion (4 elements / thread)
// ---------------------------------------------------------------------------
__global__ __launch_bounds__(256) void cvt_f32_bf16_kernel(
    const float* __restrict__ in, bf16* __restrict__ out, int n4) {
  int i = blockIdx.x * 256 + threadIdx.x;
  if (i < n4) {
    float4 f = ((const float4*)in)[i];
    bf16x4 o = {(bf16)f.x, (bf16)f.y, (bf16)f.z, (bf16)f.w};
    ((bf16x4*)out)[i] = o;
  }
}

// ---------------------------------------------------------------------------
// QKV GEMM: QKV[m][n] = sum_k X[m][k] * Wqkv[n][k]
// M=8192, N=3072, K=1024. Block = 128x256 (8 waves, 64x64 wave tile).
// Epilogue routes to Q (scaled by 1/sqrt(64)), K as (BH,T,64),
// V transposed as (BH,64,T).
// ---------------------------------------------------------------------------
__global__ __launch_bounds__(256) void qkv_gemm_kernel(
    const bf16* __restrict__ X, const bf16* __restrict__ W,
    bf16* __restrict__ Qb, bf16* __restrict__ Kb, bf16* __restrict__ Vt) {
  const int w = threadIdx.x >> 5, lane = threadIdx.x & 31;
  const int hi = lane >> 4, ln = lane & 15;
  const int m0 = blockIdx.x * 128 + (w >> 2) * 64;
  const int n0 = blockIdx.y * 256 + (w & 3) * 64;

  v8f acc[4][4] = {};
#pragma unroll 2
  for (int k = 0; k < 1024; k += 32) {
    v16bf a[4], b[4];
#pragma unroll
    for (int i = 0; i < 4; i++)
      a[i] = ldfragA(X + (size_t)(m0 + i * 16) * 1024 + k, 1024);
#pragma unroll
    for (int j = 0; j < 4; j++)
      b[j] = ldfragB(W + (size_t)(n0 + j * 16) * 1024 + k, 1024);
#pragma unroll
    for (int i = 0; i < 4; i++)
#pragma unroll
      for (int j = 0; j < 4; j++) acc[i][j] = wmma_bf16(a[i], b[j], acc[i][j]);
  }

#pragma unroll
  for (int i = 0; i < 4; i++) {
#pragma unroll
    for (int j = 0; j < 4; j++) {
      const int n = n0 + j * 16 + ln;     // output channel in [0,3072)
      const int sec = n >> 10;            // 0=q 1=k 2=v
      const int c = n & 1023;
      const int hh = c >> 6, d = c & 63;
#pragma unroll
      for (int v = 0; v < 8; v++) {
        const int g = m0 + i * 16 + v + hi * 8;  // token index
        const int t = g & 1023, bb = g >> 10;
        const int bh = bb * 16 + hh;
        const float val = acc[i][j][v];
        if (sec == 0)
          Qb[((size_t)bh * 1024 + t) * 64 + d] = (bf16)(val * 0.125f);
        else if (sec == 1)
          Kb[((size_t)bh * 1024 + t) * 64 + d] = (bf16)val;
        else
          Vt[((size_t)bh * 64 + d) * 1024 + t] = (bf16)val;
      }
    }
  }
}

// ---------------------------------------------------------------------------
// Flash attention, causal. One block = 128 q rows of one (b,h).
// 8 waves x 16 q rows, each wave loops only to its own causal limit
// (wave-private P tile in LDS; no block-wide barriers needed).
// ---------------------------------------------------------------------------
__global__ __launch_bounds__(256) void attn_kernel(
    const bf16* __restrict__ Q, const bf16* __restrict__ K,
    const bf16* __restrict__ Vt, bf16* __restrict__ Y) {
  __shared__ bf16 Plds[8][16 * 40];  // per-wave 16x32 P tile, stride 40

  const int w = threadIdx.x >> 5, lane = threadIdx.x & 31;
  const int hi = lane >> 4, ln = lane & 15;
  const int BH = blockIdx.y;
  const int bb = BH >> 4, h = BH & 15;
  const int q0 = blockIdx.x * 128 + w * 16;

  const bf16* Qp = Q + ((size_t)BH * 1024 + q0) * 64;
  const bf16* Kp = K + (size_t)BH * 1024 * 64;
  const bf16* Vp = Vt + (size_t)BH * 64 * 1024;

  v16bf qf[2];
  qf[0] = ldfragA(Qp, 64);       // d = 0..31
  qf[1] = ldfragA(Qp + 32, 64);  // d = 32..63

  v8f o[4] = {};
  float mrow[8], lrow[8];
#pragma unroll
  for (int v = 0; v < 8; v++) { mrow[v] = -INFINITY; lrow[v] = 0.f; }

  bf16* Pw = &Plds[w][0];
  // Per-wave causal bound: keys 0 .. q0+15  ->  ceil((q0+16)/32) blocks.
  const int nk = (q0 + 47) >> 5;

#pragma unroll 1
  for (int kb = 0; kb < nk; ++kb) {
    const int kk0 = kb * 32;

    // Prefetch next key block's K rows (32 rows, one per lane) and V^T rows.
    if (kb + 1 < nk) {
      const int kn = kk0 + 32;
      __builtin_prefetch(Kp + (size_t)(kn + lane) * 64, 0, 3);
      __builtin_prefetch(Vp + (size_t)lane * 1024 + kn, 0, 3);
      __builtin_prefetch(Vp + (size_t)(lane + 32) * 1024 + kn, 0, 3);
    }

    // S = Q * K^T  (16q x 32keys), two 16-col fragments
    v8f s0 = {}, s1 = {};
    s0 = wmma_bf16(qf[0], ldfragB(Kp + (size_t)kk0 * 64, 64), s0);
    s0 = wmma_bf16(qf[1], ldfragB(Kp + (size_t)kk0 * 64 + 32, 64), s0);
    s1 = wmma_bf16(qf[0], ldfragB(Kp + (size_t)(kk0 + 16) * 64, 64), s1);
    s1 = wmma_bf16(qf[1], ldfragB(Kp + (size_t)(kk0 + 16) * 64 + 32, 64), s1);

    const int key0 = kk0 + ln, key1 = kk0 + 16 + ln;
#pragma unroll
    for (int v = 0; v < 8; v++) {
      const int row = v + hi * 8;
      const int qrow = q0 + row;
      float a0 = (key0 <= qrow) ? s0[v] : -INFINITY;
      float a1 = (key1 <= qrow) ? s1[v] : -INFINITY;
      float rm = redmax16(fmaxf(a0, a1));
      const float mn = fmaxf(mrow[v], rm);
      const float alpha = __expf(mrow[v] - mn);
      const float p0 = __expf(a0 - mn);
      const float p1 = __expf(a1 - mn);
      const float rs = redsum16(p0 + p1);
      lrow[v] = lrow[v] * alpha + rs;
      mrow[v] = mn;
#pragma unroll
      for (int c = 0; c < 4; c++) o[c][v] = o[c][v] * alpha;
      Pw[row * 40 + ln] = (bf16)p0;
      Pw[row * 40 + 16 + ln] = (bf16)p1;
    }
    wave_lds_fence();  // order P stores (all lanes) before the A-frag reload

    // O += P * V  (P: 16x32 from LDS, V^T rows are d with contiguous t)
    v16bf pf = ldfragA_lds(Pw, 40);
#pragma unroll
    for (int c = 0; c < 4; c++)
      o[c] = wmma_bf16(pf, ldfragB(Vp + (size_t)(c * 16) * 1024 + kk0, 1024),
                       o[c]);
    wave_lds_fence();  // WAR: next iteration's stores stay after these reads
  }

#pragma unroll
  for (int v = 0; v < 8; v++) lrow[v] = 1.0f / lrow[v];
#pragma unroll
  for (int c = 0; c < 4; c++) {
#pragma unroll
    for (int v = 0; v < 8; v++) {
      const int row = v + hi * 8;
      const int g = bb * 1024 + q0 + row;      // token index
      const int ch = h * 64 + c * 16 + ln;     // channel
      Y[(size_t)g * 1024 + ch] = (bf16)(o[c][v] * lrow[v]);
    }
  }
}

// ---------------------------------------------------------------------------
// Output projection: out[m][n] = sum_k Y[m][k] * Wproj[n][k] + bias[n]
// M=8192, N=1024, K=1024. Same tiling as QKV GEMM, f32 output.
// ---------------------------------------------------------------------------
__global__ __launch_bounds__(256) void proj_gemm_kernel(
    const bf16* __restrict__ Yb, const bf16* __restrict__ Wp,
    const float* __restrict__ bias, float* __restrict__ out) {
  const int w = threadIdx.x >> 5, lane = threadIdx.x & 31;
  const int hi = lane >> 4, ln = lane & 15;
  const int m0 = blockIdx.x * 128 + (w >> 2) * 64;
  const int n0 = blockIdx.y * 256 + (w & 3) * 64;

  v8f acc[4][4] = {};
#pragma unroll 2
  for (int k = 0; k < 1024; k += 32) {
    v16bf a[4], b[4];
#pragma unroll
    for (int i = 0; i < 4; i++)
      a[i] = ldfragA(Yb + (size_t)(m0 + i * 16) * 1024 + k, 1024);
#pragma unroll
    for (int j = 0; j < 4; j++)
      b[j] = ldfragB(Wp + (size_t)(n0 + j * 16) * 1024 + k, 1024);
#pragma unroll
    for (int i = 0; i < 4; i++)
#pragma unroll
      for (int j = 0; j < 4; j++) acc[i][j] = wmma_bf16(a[i], b[j], acc[i][j]);
  }

#pragma unroll
  for (int j = 0; j < 4; j++) {
    const int n = n0 + j * 16 + ln;
    const float bv = bias[n];
#pragma unroll
    for (int i = 0; i < 4; i++) {
#pragma unroll
      for (int v = 0; v < 8; v++) {
        const int m = m0 + i * 16 + v + hi * 8;
        out[(size_t)m * 1024 + n] = acc[i][j][v] + bv;
      }
    }
  }
}

// ---------------------------------------------------------------------------
// Host launcher
// ---------------------------------------------------------------------------
extern "C" void kernel_launch(void* const* d_in, const int* in_sizes, int n_in,
                              void* d_out, int out_size, void* d_ws,
                              size_t ws_size, hipStream_t stream) {
  (void)in_sizes; (void)n_in; (void)out_size; (void)ws_size;
  const float* x      = (const float*)d_in[0];  // (8,1024,1024)
  const float* w_qkv  = (const float*)d_in[1];  // (3072,1024)
  const float* w_proj = (const float*)d_in[2];  // (1024,1024)
  const float* b_proj = (const float*)d_in[3];  // (1024,)
  float* out = (float*)d_out;

  // Workspace layout (bf16 elements)
  bf16* Xbf   = (bf16*)d_ws;            // 8192*1024
  bf16* Wqkvb = Xbf + 8388608;          // 3072*1024
  bf16* Wpb   = Wqkvb + 3145728;        // 1024*1024
  bf16* Qb    = Wpb + 1048576;          // 128*1024*64  (BH,T,D), pre-scaled
  bf16* Kb    = Qb + 8388608;           // 128*1024*64  (BH,T,D)
  bf16* Vt    = Kb + 8388608;           // 128*64*1024  (BH,D,T)
  bf16* Yb    = Vt + 8388608;           // 8192*1024    (token, C)

  // 1) f32 -> bf16 conversions
  cvt_f32_bf16_kernel<<<(8388608 / 4) / 256, 256, 0, stream>>>(x, Xbf,
                                                               8388608 / 4);
  cvt_f32_bf16_kernel<<<(3145728 / 4) / 256, 256, 0, stream>>>(w_qkv, Wqkvb,
                                                               3145728 / 4);
  cvt_f32_bf16_kernel<<<(1048576 / 4) / 256, 256, 0, stream>>>(w_proj, Wpb,
                                                               1048576 / 4);

  // 2) QKV projection (writes Q scaled, K, V-transposed)
  qkv_gemm_kernel<<<dim3(64, 12), 256, 0, stream>>>(Xbf, Wqkvb, Qb, Kb, Vt);

  // 3) causal flash attention -> Yb
  attn_kernel<<<dim3(8, 128), 256, 0, stream>>>(Qb, Kb, Vt, Yb);

  // 4) output projection + bias -> f32 out
  proj_gemm_kernel<<<dim3(64, 4), 256, 0, stream>>>(Yb, Wpb, b_proj, out);
}
